// ConvCrossAttention_56315611185663
// MI455X (gfx1250) — matrix operand order repaired and checked by reading, not compiled, not run on hardware
//
#include <hip/hip_runtime.h>
#include <stdint.h>

#define B_   4
#define C_   256
#define NPIX 4096   // 64*64

#define KSTR 68     // lds_k row stride (shorts): 64 + 2-dword TDM pad
#define VSTR 260    // lds_v row stride (shorts): 256 + 2-dword TDM pad
#define PSTR 66     // lds_p row stride (shorts)

typedef __attribute__((ext_vector_type(16))) __bf16        v16bf;
typedef __attribute__((ext_vector_type(8)))  float         v8f;
typedef __attribute__((ext_vector_type(8)))  unsigned int  v8u;

union Frag { v16bf bf; v8u u; };

#if defined(__has_builtin)
#if __has_builtin(__builtin_amdgcn_tensor_load_to_lds) && \
    __has_builtin(__builtin_amdgcn_s_wait_tensorcnt)
#define USE_TDM 1
#endif
#if __has_builtin(__builtin_amdgcn_global_load_async_to_lds_b64) && \
    __has_builtin(__builtin_amdgcn_s_wait_asynccnt)
#define USE_ASYNC_LDS 1
#endif
#endif

__device__ __forceinline__ unsigned short f2bf(float f) {
    unsigned int u = __float_as_uint(f);
    unsigned int r = 0x7FFFu + ((u >> 16) & 1u);
    return (unsigned short)((u + r) >> 16);
}

// A-matrix (16x32 bf16) K offset for VGPR-pair i, lane-half g (ISA 7.12.2)
__device__ __forceinline__ int a_kofs(int i, int g) {
    return ((i < 4) ? 0 : 16) + g * 8 + (i & 3) * 2;
}

// --- explicit LDS (addrspace 3) access: forces ds_load/ds_store, never FLAT.
typedef __attribute__((address_space(3))) const unsigned int   lds_cu32_t;
typedef __attribute__((address_space(3))) unsigned short       lds_u16_t;

__device__ __forceinline__ unsigned int lds_ld32(const unsigned short* p) {
    return *(lds_cu32_t*)(unsigned int)(uintptr_t)p;   // generic->LDS: addr[31:0]
}
__device__ __forceinline__ void lds_st16(unsigned short* p, unsigned short v) {
    *(lds_u16_t*)(unsigned int)(uintptr_t)p = v;
}

#ifdef USE_TDM
typedef __attribute__((ext_vector_type(4))) unsigned int v4u;
typedef __attribute__((ext_vector_type(8))) int          v8i;
typedef __attribute__((ext_vector_type(4))) int          v4i;

// Issue one 2D TENSOR_LOAD_TO_LDS (D# per ISA 8.3/8.4).  Dims/strides in
// 2-byte elements; LDS pad: pad_amount(code) dwords per pad_interval(code).
__device__ __forceinline__ void tdm_load_2d(const void* gaddr, void* laddr,
                                            unsigned tensor_d0, unsigned tensor_d1,
                                            unsigned tile_d0, unsigned tile_d1,
                                            unsigned long long stride0,
                                            unsigned pad_interval, unsigned pad_amount)
{
    unsigned long long ga = (unsigned long long)(uintptr_t)gaddr;
    unsigned la = (unsigned)(uintptr_t)laddr;      // generic->LDS: addr[31:0]

    v4u g0;
    g0[0] = 1u;                                    // count=1, user descriptor
    g0[1] = la;                                    // lds_addr (bytes)
    g0[2] = (unsigned)(ga & 0xFFFFFFFFu);          // global_addr[31:0]
    g0[3] = (unsigned)((ga >> 32) & 0x1FFFFFFu)    // global_addr[56:32]
          | (2u << 30);                            // type = 2 ("image")

    v8i g1;
    g1[0] = (int)((1u << 16)                       // data_size: 2 bytes
          | (1u << 20)                             // pad_enable
          | (pad_interval << 22) | (pad_amount << 25));
    g1[1] = (int)((tensor_d0 & 0xFFFFu) << 16);    // [63:48] tensor_dim0 lo16
    g1[2] = (int)(((tensor_d0 >> 16) & 0xFFFFu) | ((tensor_d1 & 0xFFFFu) << 16));
    g1[3] = (int)(((tensor_d1 >> 16) & 0xFFFFu) | ((tile_d0 & 0xFFFFu) << 16));
    g1[4] = (int)(tile_d1 & 0xFFFFu);              // tile_dim1; tile_dim2 = 0
    g1[5] = (int)(stride0 & 0xFFFFFFFFull);        // tensor_dim0_stride lo32
    g1[6] = (int)((stride0 >> 32) & 0xFFFFu);      // stride hi16; dim1_stride=0
    g1[7] = 0;

    const v4i gz4 = {};
    const v8i gz8 = {};
    __builtin_amdgcn_tensor_load_to_lds(g0, g1, gz4, gz4, gz8, 0);
}
#endif

#ifdef USE_ASYNC_LDS
typedef __attribute__((ext_vector_type(2))) int          v2i_t;
typedef __attribute__((address_space(1))) v2i_t          g_v2i;
typedef __attribute__((address_space(3))) v2i_t          l_v2i;

__device__ __forceinline__ void async_b64(const void* g, void* l) {
    __builtin_amdgcn_global_load_async_to_lds_b64(
        (g_v2i*)(uintptr_t)g,
        (l_v2i*)(unsigned int)(uintptr_t)l, 0, 0);
}
#endif

// Stage one 64-key tile of K^T ([256][64] -> KSTR) and V ([64][256] -> VSTR).
// TDM path: 2 tensor DMAs (call from ONE wave).  Async/sync: all threads.
__device__ __forceinline__ void issue_tile(const unsigned short* __restrict__ Kb,
                                           const unsigned short* __restrict__ Vb,
                                           unsigned short* ldsk, unsigned short* ldsv,
                                           int kv0, int tid)
{
#ifdef USE_TDM
    // K^T: tile 64 x 256 rows, row stride 4096; pad 2 dw per 32 dw (KSTR=68)
    tdm_load_2d(&Kb[kv0], ldsk, NPIX, C_, 64, 256, (unsigned long long)NPIX, 4u, 1u);
    // V: tile 256 x 64 rows, row stride 256; pad 2 dw per 128 dw (VSTR=260)
    tdm_load_2d(&Vb[(size_t)kv0 * C_], ldsv, C_, NPIX, 256, 64,
                (unsigned long long)C_, 6u, 1u);
    (void)tid;
#elif defined(USE_ASYNC_LDS)
    #pragma unroll
    for (int it = 0; it < 16; ++it) {
        int d = it * 256 + tid;
        int c = d >> 4, q = (d & 15) * 4;
        async_b64(&Kb[(size_t)c * NPIX + kv0 + q], &ldsk[c * KSTR + q]);
    }
    #pragma unroll
    for (int it = 0; it < 16; ++it) {
        int d = it * 256 + tid;
        int j = d >> 6, q = (d & 63) * 4;
        async_b64(&Vb[(size_t)(kv0 + j) * C_ + q], &ldsv[j * VSTR + q]);
    }
#else
    for (int d = tid; d < 4096; d += 256) {
        int c = d >> 4, q = (d & 15) * 4;
        const uint2 x = *(const uint2*)&Kb[(size_t)c * NPIX + kv0 + q];
        *(unsigned int*)&ldsk[c * KSTR + q]     = x.x;
        *(unsigned int*)&ldsk[c * KSTR + q + 2] = x.y;
    }
    for (int d = tid; d < 4096; d += 256) {
        int j = d >> 6, q = (d & 63) * 4;
        const uint2 x = *(const uint2*)&Vb[(size_t)(kv0 + j) * C_ + q];
        *(unsigned int*)&ldsv[j * VSTR + q]     = x.x;
        *(unsigned int*)&ldsv[j * VSTR + q + 2] = x.y;
    }
#endif
}

// ---------------------------------------------------------------------------
// Kernel 1: 1x1-conv projection  out[b,n,o] = sum_c x[b,c,n] * W[o,c] + bias[o]
// ---------------------------------------------------------------------------
template <int TR>
__global__ __launch_bounds__(256) void proj_qkv(const float* __restrict__ x,
                                                const float* __restrict__ Wt,
                                                const float* __restrict__ bias,
                                                unsigned short* __restrict__ out)
{
    __shared__ unsigned short lds_x[128 * 34];  // [pixel][c]
    __shared__ unsigned short lds_w[32 * 34];   // [o][c]

    const int tid  = threadIdx.x;
    const int lane = tid & 31;
    const int wave = tid >> 5;
    const int g    = lane >> 4;
    const int m    = lane & 15;

    const int p0 = blockIdx.x * 128;
    const int o0 = blockIdx.y * 32;
    const int b  = blockIdx.z;

    const float* xb = x + (size_t)b * C_ * NPIX;

    const v8f vzero = {};
    v8f acc[2];
    acc[0] = vzero;
    acc[1] = vzero;

    for (int kc = 0; kc < C_; kc += 32) {
        for (int idx = tid; idx < 32 * 128; idx += 256) {
            int c = idx >> 7;
            int p = idx & 127;
            lds_st16(&lds_x[p * 34 + c], f2bf(xb[(size_t)(kc + c) * NPIX + p0 + p]));
        }
        for (int idx = tid; idx < 32 * 32; idx += 256) {
            int o = idx >> 5;
            int c = idx & 31;
            lds_st16(&lds_w[o * 34 + c], f2bf(Wt[(size_t)(o0 + o) * C_ + kc + c]));
        }
        __syncthreads();

        Frag a;
        const int arow = (wave * 16 + m) * 34;
        #pragma unroll
        for (int i = 0; i < 8; ++i)
            a.u[i] = lds_ld32(&lds_x[arow + a_kofs(i, g)]);

        Frag b0, b1;
        #pragma unroll
        for (int i = 0; i < 8; ++i) {
            b0.u[i] = lds_ld32(&lds_w[m * 34 + a_kofs(i, g)]);
            b1.u[i] = lds_ld32(&lds_w[(16 + m) * 34 + a_kofs(i, g)]);
        }
        acc[0] = __builtin_amdgcn_wmma_f32_16x16x32_bf16(false, a.bf, false, b0.bf,
                                                         (short)0, acc[0], false, false);
        acc[1] = __builtin_amdgcn_wmma_f32_16x16x32_bf16(false, a.bf, false, b1.bf,
                                                         (short)0, acc[1], false, false);
        __syncthreads();
    }

    const int prow = p0 + wave * 16;
    #pragma unroll
    for (int og = 0; og < 2; ++og) {
        const int oc = o0 + og * 16 + m;
        const float bv = bias[oc];
        #pragma unroll
        for (int r = 0; r < 8; ++r) {
            const int row = prow + r + 8 * g;
            const float val = acc[og][r] + bv;
            size_t idx = TR ? ((size_t)b * C_ + oc) * NPIX + row
                            : ((size_t)b * NPIX + row) * C_ + oc;
            out[idx] = f2bf(val);
        }
    }
}

// ---------------------------------------------------------------------------
// Kernel 2: flash attention, double-buffered K/V tiles staged by the TDM
// (tensor_load_to_lds) or async-to-LDS loads.
// ---------------------------------------------------------------------------
__global__ __launch_bounds__(256) void flash_attn(const unsigned short* __restrict__ Q,
                                                  const unsigned short* __restrict__ Kt,
                                                  const unsigned short* __restrict__ V,
                                                  float* __restrict__ out)
{
    extern __shared__ unsigned short smem[];
    unsigned short* lds_k[2];
    unsigned short* lds_v[2];
    lds_k[0] = smem;                                   // [256][KSTR]
    lds_k[1] = lds_k[0] + 256 * KSTR;
    lds_v[0] = lds_k[1] + 256 * KSTR;                  // [64][VSTR]
    lds_v[1] = lds_v[0] + 64 * VSTR;
    unsigned short* lds_p = lds_v[1] + 64 * VSTR;      // [8 waves][16][PSTR]

    const int tid  = threadIdx.x;
    const int lane = tid & 31;
    const int wave = tid >> 5;
    const int g    = lane >> 4;
    const int m    = lane & 15;

    const int b  = blockIdx.y;
    const int q0 = blockIdx.x * 128 + wave * 16;

    const unsigned short* Qb = Q  + (size_t)b * NPIX * C_;
    const unsigned short* Kb = Kt + (size_t)b * C_ * NPIX;
    const unsigned short* Vb = V  + (size_t)b * NPIX * C_;

    // Preload this wave's Q rows as 8 A-fragments (16 rows x 256 channels)
    Frag qa[8];
    #pragma unroll
    for (int t = 0; t < 8; ++t)
        #pragma unroll
        for (int i = 0; i < 8; ++i)
            qa[t].u[i] = *(const unsigned int*)
                &Qb[(size_t)(q0 + m) * C_ + t * 32 + a_kofs(i, g)];

    const v8f vzero = {};
    v8f acc[16];
    #pragma unroll
    for (int t = 0; t < 16; ++t) acc[t] = vzero;
    float mrow[8], lrow[8];
    #pragma unroll
    for (int r = 0; r < 8; ++r) { mrow[r] = -1e30f; lrow[r] = 0.f; }

    const float sc2 = 0.0625f * 1.44269504088896f;  // 1/sqrt(C) * log2(e)
    const int   nT  = NPIX / 64;

    // prologue: stage tile 0 into buffer 0
#ifdef USE_TDM
    if (wave == 0) issue_tile(Kb, Vb, lds_k[0], lds_v[0], 0, tid);
#else
    issue_tile(Kb, Vb, lds_k[0], lds_v[0], 0, tid);
#endif

    for (int kt = 0; kt < nT; ++kt) {
        const int cur = kt & 1;

#ifdef USE_TDM
        if (wave == 0) {
            if (kt + 1 < nT) {
                issue_tile(Kb, Vb, lds_k[1 - cur], lds_v[1 - cur], (kt + 1) * 64, tid);
                // 2 newest TDM ops belong to tile kt+1; in-order completion
                // per wave => tile kt is resident.
                __builtin_amdgcn_s_wait_tensorcnt(2);
            } else {
                __builtin_amdgcn_s_wait_tensorcnt(0);
            }
        }
#else
        if (kt + 1 < nT)
            issue_tile(Kb, Vb, lds_k[1 - cur], lds_v[1 - cur], (kt + 1) * 64, tid);
#ifdef USE_ASYNC_LDS
        if (kt + 1 < nT) __builtin_amdgcn_s_wait_asynccnt(32);
        else             __builtin_amdgcn_s_wait_asynccnt(0);
#endif
#endif
        // warm L2 for tile kt+2 (staging covers kt+1)
        if (kt + 2 < nT) {
            __builtin_prefetch(&Kb[(size_t)tid * NPIX + (kt + 2) * 64], 0, 1);
            __builtin_prefetch(&Vb[(size_t)((kt + 2) * 64 + (tid >> 2)) * C_ + (tid & 3) * 64], 0, 1);
        }
        __syncthreads();

        const unsigned short* ldsk = lds_k[cur];
        const unsigned short* ldsv = lds_v[cur];

        // S = Q * K^T  (16 x 64): 4 column groups x 8 K-steps
        v8f S[4];
        #pragma unroll
        for (int kvg = 0; kvg < 4; ++kvg) {
            v8f s = vzero;
            #pragma unroll
            for (int t = 0; t < 8; ++t) {
                Frag kb;
                #pragma unroll
                for (int i = 0; i < 8; ++i)
                    kb.u[i] = (unsigned int)ldsk[(t * 32 + a_kofs(i, g)) * KSTR + kvg * 16 + m]
                            | ((unsigned int)ldsk[(t * 32 + a_kofs(i, g) + 1) * KSTR + kvg * 16 + m] << 16);
                s = __builtin_amdgcn_wmma_f32_16x16x32_bf16(false, qa[t].bf, false, kb.bf,
                                                            (short)0, s, false, false);
            }
            S[kvg] = s;
        }

        // online softmax (exp2 domain); row stats live in one lane half
        #pragma unroll
        for (int r = 0; r < 8; ++r) {
            float t0 = fmaxf(fmaxf(S[0][r], S[1][r]), fmaxf(S[2][r], S[3][r])) * sc2;
            #pragma unroll
            for (int msk = 8; msk >= 1; msk >>= 1)
                t0 = fmaxf(t0, __shfl_xor(t0, msk, 32));
            const float mnew  = fmaxf(mrow[r], t0);
            const float alpha = exp2f(mrow[r] - mnew);
            mrow[r] = mnew;
            float rs = 0.f;
            #pragma unroll
            for (int kvg = 0; kvg < 4; ++kvg) {
                const float p = exp2f(S[kvg][r] * sc2 - mnew);
                S[kvg][r] = p;
                rs += p;
            }
            #pragma unroll
            for (int msk = 8; msk >= 1; msk >>= 1)
                rs += __shfl_xor(rs, msk, 32);
            lrow[r] = lrow[r] * alpha + rs;
            #pragma unroll
            for (int t = 0; t < 16; ++t) acc[t][r] *= alpha;
        }

        // write P (bf16) to per-wave LDS scratch, D-layout -> row-major [16][64]
        unsigned short* pw = lds_p + wave * (16 * PSTR);
        #pragma unroll
        for (int kvg = 0; kvg < 4; ++kvg)
            #pragma unroll
            for (int r = 0; r < 8; ++r)
                lds_st16(&pw[(r + 8 * g) * PSTR + kvg * 16 + m], f2bf(S[kvg][r]));

        // O += P * V  (two 32-deep K steps over the 64 keys)
        #pragma unroll
        for (int h = 0; h < 2; ++h) {
            Frag pa;
            #pragma unroll
            for (int i = 0; i < 8; ++i)
                pa.u[i] = lds_ld32(&pw[m * PSTR + h * 32 + a_kofs(i, g)]);
            #pragma unroll
            for (int cg = 0; cg < 16; ++cg) {
                Frag vb;
                #pragma unroll
                for (int i = 0; i < 8; ++i)
                    vb.u[i] = (unsigned int)ldsv[(h * 32 + a_kofs(i, g)) * VSTR + cg * 16 + m]
                            | ((unsigned int)ldsv[(h * 32 + a_kofs(i, g) + 1) * VSTR + cg * 16 + m] << 16);
                acc[cg] = __builtin_amdgcn_wmma_f32_16x16x32_bf16(false, pa.bf, false, vb.bf,
                                                                  (short)0, acc[cg], false, false);
            }
        }
        __syncthreads();   // tile kt buffer may be overwritten next iteration
    }

    // normalize by softmax denominator; store f32, out layout [B][C][N]
    #pragma unroll
    for (int r = 0; r < 8; ++r) {
        const float inv = 1.0f / lrow[r];
        const int row = q0 + r + 8 * g;
        #pragma unroll
        for (int cg = 0; cg < 16; ++cg) {
            const int c = cg * 16 + m;
            out[((size_t)b * C_ + c) * NPIX + row] = acc[cg][r] * inv;
        }
    }
}

// ---------------------------------------------------------------------------
extern "C" void kernel_launch(void* const* d_in, const int* in_sizes, int n_in,
                              void* d_out, int out_size, void* d_ws, size_t ws_size,
                              hipStream_t stream) {
    const float* x1 = (const float*)d_in[0];
    const float* x2 = (const float*)d_in[1];
    const float* Wq = (const float*)d_in[2];
    const float* bq = (const float*)d_in[3];
    const float* Wk = (const float*)d_in[4];
    const float* bk = (const float*)d_in[5];
    const float* Wv = (const float*)d_in[6];
    const float* bv = (const float*)d_in[7];

    // bf16 Q/K/V in workspace: 3 * 4*4096*256*2B = 24 MB
    unsigned short* Qw = (unsigned short*)d_ws;
    unsigned short* Kw = Qw + (size_t)B_ * NPIX * C_;
    unsigned short* Vw = Kw + (size_t)B_ * C_ * NPIX;

    dim3 gp(NPIX / 128, C_ / 32, B_);   // (32, 8, 4)
    proj_qkv<0><<<gp, 256, 0, stream>>>(x1, Wq, bq, Qw);  // Q: [B][N][C]
    proj_qkv<1><<<gp, 256, 0, stream>>>(x2, Wk, bk, Kw);  // K: [B][C][N]
    proj_qkv<0><<<gp, 256, 0, stream>>>(x2, Wv, bv, Vw);  // V: [B][N][C]

    dim3 ga(NPIX / 128, B_);            // (32, 4)
    size_t shmem = (size_t)(2 * 256 * KSTR + 2 * 64 * VSTR + 8 * 16 * PSTR)
                   * sizeof(unsigned short);
    flash_attn<<<ga, 256, shmem, stream>>>(Qw, Kw, Vw, (float*)d_out);
}
